// DigitCaps_3822520893884
// MI455X (gfx1250) — compile-verified, matrix operand order checked by reading, not run on hardware
//
#include <hip/hip_runtime.h>
#include <hip/hip_bf16.h>
#include <math.h>

// ---------------------------------------------------------------------------
// DigitCaps dynamic routing, MI455X (gfx1250, wave32, WMMA).
//
// u is NEVER materialized (would be 377MB of HBM traffic, ~2.6GB moved over
// 7 touches). U[16b x 16w] tiles are recomputed per routing pass with
// V_WMMA_F32_16X16X4_F32 (exact f32, K=8 via two chained x4 WMMAs).
// Logits after iter k equal u·(v1+...+vk), so one running v-sum buffer
// suffices -> exactly 3 fused passes. HBM traffic ~= x(18.9MB) + W(5.9MB),
// both L2-resident (192MB L2); recompute passes are fed from L2/L0.
// ---------------------------------------------------------------------------

typedef float    v2f  __attribute__((ext_vector_type(2)));
typedef float    v8f  __attribute__((ext_vector_type(8)));
typedef _Float16 v16h __attribute__((ext_vector_type(16)));

#define NB 512
#define NO 10
#define NI 1152
#define NW 16
#define NS 8
#define BT_COUNT 32                 // 512 / 16 batch tiles
#define IC_COUNT 8                  // i-chunks (grid dim)
#define WAVES    4                  // waves per WG (128 threads)
#define I_PER_WAVE (NI / (IC_COUNT * WAVES))   // 36
#define WOSTRIDE (NI * NW * NS)     // 147456 floats between o-slices of W
#define VSLICE   (16 * NO * NW)     // 2560 floats of v per batch tile
#define SPART_STRIDE VSLICE
#define SPART_SLICE  (BT_COUNT * SPART_STRIDE) // 81920 floats per ic

// U[16x16] += A[16x8] * B[8x16], A from x rows, B from W (s innermost).
// f32 path: two V_WMMA_F32_16X16X4_F32. ISA layouts:
//   A 16x4 : lanes0-15 {K0,K1}, lanes16-31 {K2,K3}  -> float2 @ s=klo
//   B 4x16 : VGPR0 K=(lane<16?0:2), VGPR1 K+1       -> float2 @ s=klo
//   C/D    : VGPR r = rows (r, r+8), lane&15 = column
__device__ __forceinline__ v8f wmma_u(const float* __restrict__ xrow,
                                      const float* __restrict__ wrow,
                                      int klo, int lane, v8f acc) {
#if defined(__HIP_DEVICE_COMPILE__)
#if __has_builtin(__builtin_amdgcn_wmma_f32_16x16x4_f32)
  v2f al = *(const v2f*)(xrow + klo);
  v2f ah = *(const v2f*)(xrow + 4 + klo);
  v2f bl = *(const v2f*)(wrow + klo);
  v2f bh = *(const v2f*)(wrow + 4 + klo);
  acc = __builtin_amdgcn_wmma_f32_16x16x4_f32(false, al, false, bl,
                                              (short)0, acc, false, false);
  acc = __builtin_amdgcn_wmma_f32_16x16x4_f32(false, ah, false, bh,
                                              (short)0, acc, false, false);
#else
  // Fallback: f16 16x16x32, K=0..7 carries data, rest zero.
  v16h A = {}, Bv = {};
  if (lane < 16) {
#pragma unroll
    for (int e = 0; e < 8; ++e) {
      A[e]  = (_Float16)xrow[e];
      Bv[e] = (_Float16)wrow[e];
    }
  }
  acc = __builtin_amdgcn_wmma_f32_16x16x32_f16(false, A, false, Bv,
                                               (short)0, acc, false, false);
  (void)klo;
#endif
#else
  (void)xrow; (void)wrow; (void)klo; (void)lane;
#endif
  return acc;
}

// One fused routing pass. PHASE==0: uniform c=1/10 (iter 1).
// PHASE==1: logits = U · vbuf (vbuf = v1 for iter2, v1+v2 for iter3).
template <int PHASE>
__global__ __launch_bounds__(128)
void caps_pass(const float* __restrict__ x, const float* __restrict__ W,
               const float* __restrict__ vbuf, float* __restrict__ s_part) {
  __shared__ float s_red[WAVES][SPART_STRIDE];   // 40KB
  __shared__ float v_sh[VSLICE];                 // 10KB (phase>0)

  const int lane  = threadIdx.x & 31;
  const int wave  = threadIdx.x >> 5;
  const int bt    = blockIdx.x & (BT_COUNT - 1);
  const int ic    = blockIdx.x >> 5;
  const int wcol  = lane & 15;       // N column (w) / M row (batch) index
  const int whalf = lane >> 4;
  const int klo   = whalf * 2;       // K slot for this lane half

  // Stage this batch tile's v slice (2560 floats) into LDS once.
  if (PHASE > 0) {
    const float4* vsrc = (const float4*)(vbuf + (size_t)bt * VSLICE);
    float4* vdst = (float4*)v_sh;
    for (int e = threadIdx.x; e < VSLICE / 4; e += 128) vdst[e] = vsrc[e];
    __syncthreads();
  }

  const float* xrow_base = x + (size_t)(bt * 16 + wcol) * (NI * NS);
  const int vls_base = (8 * whalf) * (NO * NW) + wcol;  // LDS v index, r=o=0

  float sacc[NO][8];
#pragma unroll
  for (int o = 0; o < NO; ++o)
#pragma unroll
    for (int r = 0; r < 8; ++r) sacc[o][r] = 0.0f;

  const int i0 = ic * (NI / IC_COUNT) + wave * I_PER_WAVE;

  for (int ii = 0; ii < I_PER_WAVE; ++ii) {
    const int i = i0 + ii;
    const float* xrow  = xrow_base + i * NS;
    const float* wbase = W + (size_t)(i * NW + wcol) * NS;  // o=0 slice
    __builtin_prefetch(xrow + NS, 0, 3);        // next x row, WGP scope
    __builtin_prefetch(wbase + NW * NS, 0, 3);  // next i's W, WGP scope

    float cw[8][NO];
    if (PHASE == 0) {
#pragma unroll
      for (int r = 0; r < 8; ++r)
#pragma unroll
        for (int o = 0; o < NO; ++o) cw[r][o] = 0.1f;
    } else {
      float lg[8][NO];
#pragma unroll
      for (int o = 0; o < NO; ++o) {
        v8f U = {};
        // o-offset is a constant -> folds into the load IOFFSET
        U = wmma_u(xrow, wbase + (size_t)o * WOSTRIDE, klo, lane, U);
#pragma unroll
        for (int r = 0; r < 8; ++r) {
          // agreement a[b,o] = sum_w U[b,w]*v[b,o,w]; v via ds_load imm-offset
          float p = U[r] * v_sh[vls_base + o * NW + r * (NO * NW)];
          p += __shfl_xor(p, 1);
          p += __shfl_xor(p, 2);
          p += __shfl_xor(p, 4);
          p += __shfl_xor(p, 8);
          lg[r][o] = p;
        }
      }
      // lane-local softmax over O for each of the 8 batch rows
#pragma unroll
      for (int r = 0; r < 8; ++r) {
        float m = lg[r][0];
#pragma unroll
        for (int o = 1; o < NO; ++o) m = fmaxf(m, lg[r][o]);
        float ssum = 0.0f;
#pragma unroll
        for (int o = 0; o < NO; ++o) {
          float e = __expf(lg[r][o] - m);
          lg[r][o] = e;
          ssum += e;
        }
        float inv = 1.0f / ssum;
#pragma unroll
        for (int o = 0; o < NO; ++o) cw[r][o] = lg[r][o] * inv;
      }
    }

    // s[b,o,w] += c[b,o] * U[b,w]  (recompute U; W pairs hit L0)
#pragma unroll
    for (int o = 0; o < NO; ++o) {
      v8f U = {};
      U = wmma_u(xrow, wbase + (size_t)o * WOSTRIDE, klo, lane, U);
#pragma unroll
      for (int r = 0; r < 8; ++r) sacc[o][r] += cw[r][o] * U[r];
    }
  }

  // wave partials -> LDS -> per-WG slice in workspace (fixed order, no atomics)
#pragma unroll
  for (int o = 0; o < NO; ++o)
#pragma unroll
    for (int r = 0; r < 8; ++r) {
      int b = r + 8 * whalf;
      s_red[wave][(b * NO + o) * NW + wcol] = sacc[o][r];
    }
  __syncthreads();
  float4* dst4 = (float4*)(s_part + (size_t)(ic * BT_COUNT + bt) * SPART_STRIDE);
  const float4* r0 = (const float4*)s_red[0];
  const float4* r1 = (const float4*)s_red[1];
  const float4* r2 = (const float4*)s_red[2];
  const float4* r3 = (const float4*)s_red[3];
  for (int e = threadIdx.x; e < SPART_STRIDE / 4; e += 128) {
    float4 a = r0[e], b = r1[e], c = r2[e], d = r3[e];
    float4 o4;
    o4.x = a.x + b.x + c.x + d.x;
    o4.y = a.y + b.y + c.y + d.y;
    o4.z = a.z + b.z + c.z + d.z;
    o4.w = a.w + b.w + c.w + d.w;
    dst4[e] = o4;
  }
}

// Reduce the 8 i-chunk partials, squash, emit v.
// mode 0: vbuf = v1 ; mode 1: vbuf += v (keep v1+v2) ; mode 2: out = v.
__global__ __launch_bounds__(256)
void caps_squash(const float* __restrict__ s_part, float* __restrict__ vbuf,
                 float* __restrict__ out, int mode) {
  const int t = blockIdx.x * 256 + threadIdx.x;  // 81920 = 512*10*16
  const int w = t & 15;
  const int o = (t >> 4) % NO;
  const int b = t / (NO * NW);
  const int base = (b >> 4) * SPART_STRIDE + ((b & 15) * NO + o) * NW + w;
  float s = 0.0f;
#pragma unroll
  for (int sl = 0; sl < IC_COUNT; ++sl) s += s_part[sl * SPART_SLICE + base];
  float sq = s * s;          // ||s||^2 over w: 16-lane half reduction
  sq += __shfl_xor(sq, 1);
  sq += __shfl_xor(sq, 2);
  sq += __shfl_xor(sq, 4);
  sq += __shfl_xor(sq, 8);
  float scale = (sq > 0.0f) ? (sqrtf(sq) / (1.0f + sq)) : 0.0f;
  float v = s * scale;
  if (mode == 0)       vbuf[t] = v;
  else if (mode == 1)  vbuf[t] = vbuf[t] + v;   // running v-sum for logits
  else                 out[t] = v;
}

extern "C" void kernel_launch(void* const* d_in, const int* in_sizes, int n_in,
                              void* d_out, int out_size, void* d_ws,
                              size_t ws_size, hipStream_t stream) {
  (void)in_sizes; (void)n_in; (void)out_size; (void)ws_size;
  const float* x = (const float*)d_in[0];   // [512,32,8,6,6] == [512,1152,8]
  const float* W = (const float*)d_in[1];   // [1,10,1152,16,8]
  float* out = (float*)d_out;               // [512,10,16]
  float* s_part = (float*)d_ws;                            // 2.62 MB
  float* vbuf = s_part + (size_t)IC_COUNT * SPART_SLICE;   // 327 KB

  dim3 gA(BT_COUNT * IC_COUNT), bA(128);
  dim3 gB((NB * NO * NW) / 256), bB(256);

  // iter 1: uniform c -> s1 -> v1
  caps_pass<0><<<gA, bA, 0, stream>>>(x, W, nullptr, s_part);
  caps_squash<<<gB, bB, 0, stream>>>(s_part, vbuf, out, 0);
  // iter 2: logits = u·v1 -> s2 -> v2 ; vbuf becomes v1+v2
  caps_pass<1><<<gA, bA, 0, stream>>>(x, W, vbuf, s_part);
  caps_squash<<<gB, bB, 0, stream>>>(s_part, vbuf, out, 1);
  // iter 3: logits = u·(v1+v2) -> s3 -> v3 = output
  caps_pass<1><<<gA, bA, 0, stream>>>(x, W, vbuf, s_part);
  caps_squash<<<gB, bB, 0, stream>>>(s_part, vbuf, out, 2);
}
